// RNN_43267500540088
// MI455X (gfx1250) — compile-verified, matrix operand order checked by reading, not run on hardware
//
#include <hip/hip_runtime.h>
#include <hip/hip_bf16.h>
#include <math.h>

// B=2048, L=256, F=64, H=32, K=2
#define LL 256
#define FF 64
#define HHID 32
#define ROWS 16   // batch rows per workgroup (WMMA M)
#define NT 4      // 16-col tiles per wave -> 64 cols = one gate per wave

typedef __attribute__((ext_vector_type(16))) _Float16 v16h;
typedef __attribute__((ext_vector_type(8)))  _Float16 v8h;
typedef __attribute__((ext_vector_type(8)))  float    v8f;

union AF { v16h v; v8h h[2]; };

__device__ __forceinline__ float fsig(float x)  { return 1.0f / (1.0f + __expf(-x)); }
__device__ __forceinline__ float ftanh(float x) { return 1.0f - 2.0f / (1.0f + __expf(2.0f * x)); }

__global__ __launch_bounds__(128) void lstm_logp_kernel(
    const int*   __restrict__ s,     const float* __restrict__ g,
    const float* __restrict__ W_emb, const float* __restrict__ b_emb,
    const float* __restrict__ W_g1,  const float* __restrict__ b_g1,
    const float* __restrict__ W_g2,  const float* __restrict__ b_g2,
    const float* __restrict__ W_gh,  const float* __restrict__ b_gh,
    const float* __restrict__ W_gc,  const float* __restrict__ b_gc,
    const float* __restrict__ Wi,    const float* __restrict__ Wh,
    const float* __restrict__ b_lstm,
    const float* __restrict__ W_amp, const float* __restrict__ b_amp,
    float* __restrict__ out)
{
    // zbuf: per-step gate outputs (4 gates x 16 rows x 64 feats, f32).
    // Before the time loop the same 16KB is reused as base[16][256].
    __shared__ __attribute__((aligned(16))) float    zbuf[4][ROWS][FF];
    __shared__ __attribute__((aligned(16))) _Float16 hbuf[ROWS][FF];     // h state, f16, A-matrix source
    __shared__ __attribute__((aligned(16))) float    xrow[2][4 * FF];    // token-dependent input rows
    __shared__ float tanhbuf[ROWS][HHID];
    __shared__ float gvecbuf[ROWS][FF];
    __shared__ float gsh[ROWS];

    const int tid  = threadIdx.x;
    const int wave = tid >> 5;        // 0..3 : gate index
    const int lane = tid & 31;
    const int hl   = lane >> 4;       // lane half (A/B fragment layout)
    const int lm   = lane & 15;
    const int row0 = blockIdx.x * ROWS;

    // ---------------- one-time setup ----------------
    if (tid < ROWS) gsh[tid] = g[row0 + tid];
    __syncthreads();

    // tanh(g*W_g1 + b_g1) : [16][32]
    for (int i = tid; i < ROWS * HHID; i += 128) {
        int m = i >> 5, hh = i & 31;
        tanhbuf[m][hh] = ftanh(gsh[m] * W_g1[hh] + b_g1[hh]);
    }
    __syncthreads();
    // g_vec[m][f]
    for (int i = tid; i < ROWS * FF; i += 128) {
        int m = i >> 6, f = i & 63;
        float acc = b_g2[f];
        #pragma unroll 8
        for (int hh = 0; hh < HHID; ++hh) acc += tanhbuf[m][hh] * W_g2[hh * FF + f];
        gvecbuf[m][f] = acc;
    }
    __syncthreads();
    // base[m][n] = g_vec[m] @ Wi   (into zbuf viewed as [16][256])
    float (*baseLDS)[4 * FF] = (float (*)[4 * FF]) & zbuf[0][0][0];
    for (int i = tid; i < ROWS * 4 * FF; i += 128) {
        int m = i >> 8, n = i & 255;
        float acc = 0.f;
        #pragma unroll 8
        for (int f = 0; f < FF; ++f) acc += gvecbuf[m][f] * Wi[f * (4 * FF) + n];
        baseLDS[m][n] = acc;
    }
    // xrow[k][n] = (W_emb[k]+b_emb) @ Wi + b_lstm
    for (int i = tid; i < 2 * 4 * FF; i += 128) {
        int k = i >> 8, n = i & 255;
        float acc = b_lstm[n];
        #pragma unroll 8
        for (int f = 0; f < FF; ++f) acc += (W_emb[k * FF + f] + b_emb[f]) * Wi[f * (4 * FF) + n];
        xrow[k][n] = acc;
    }
    __syncthreads();

    // ---- resident register fragments (per wave) ----
    // B fragments of Wh (f32 -> f16) for this wave's 64 output columns.
    // 16-bit B layout (mirror of ISA A layout): lane<16: n=lm, halves K {8hl..8hl+7, +16}.
    v16h bfrag[NT][2];
    const int ncol0 = wave * 64;
    #pragma unroll
    for (int jn = 0; jn < NT; ++jn) {
        #pragma unroll
        for (int kc = 0; kc < 2; ++kc) {
            AF bf;
            const int col = ncol0 + jn * 16 + lm;
            const int kb  = kc * 32 + hl * 8;
            #pragma unroll
            for (int i = 0; i < 8; ++i) bf.h[0][i] = (_Float16)Wh[(kb + i) * (4 * FF) + col];
            #pragma unroll
            for (int i = 0; i < 8; ++i) bf.h[1][i] = (_Float16)Wh[(kb + 16 + i) * (4 * FF) + col];
            bfrag[jn][kc] = bf.v;
        }
    }
    // base accumulator fragments (C layout: vgpr i -> row i+8*hl, col lm)
    v8f basefrag[NT];
    #pragma unroll
    for (int jn = 0; jn < NT; ++jn) {
        v8f bf;
        #pragma unroll
        for (int i = 0; i < 8; ++i) bf[i] = baseLDS[i + 8 * hl][ncol0 + jn * 16 + lm];
        basefrag[jn] = bf;
    }

    // ---- per-thread elementwise state: row r, features fb..fb+7 ----
    const int r  = tid >> 3;
    const int fb = (tid & 7) * 8;
    float creg[8], wamp0[8], wamp1[8];
    {
        const float gv = gsh[r];
        #pragma unroll
        for (int i = 0; i < 8; ++i) {
            const int f = fb + i;
            creg[i]    = gv * W_gc[f] + b_gc[f];                 // c0
            hbuf[r][f] = (_Float16)(gv * W_gh[f] + b_gh[f]);     // h0
            wamp0[i]   = W_amp[f * 2 + 0];
            wamp1[i]   = W_amp[f * 2 + 1];
        }
    }
    const float ba0 = b_amp[0], ba1 = b_amp[1];
    float acc_lp = 0.f;
    const int* srow = s + (size_t)(row0 + r) * LL;
    int tok_in = 0;   // s_pad[t] ; carried register: next input token == this step's target

    // ---------------- time loop ----------------
    for (int t = 0; t < LL; ++t) {
        __syncthreads();   // hbuf from previous step ready; zbuf free

        // A fragments of h (16-bit A 16x32 layout), two K-chunks
        AF a0, a1;
        a0.h[0] = *(const v8h*)&hbuf[lm][hl * 8];
        a0.h[1] = *(const v8h*)&hbuf[lm][hl * 8 + 16];
        a1.h[0] = *(const v8h*)&hbuf[lm][32 + hl * 8];
        a1.h[1] = *(const v8h*)&hbuf[lm][32 + hl * 8 + 16];

        // target token for this step; also next step's input token
        const int tok = srow[t];

        // z_gate = base + h @ Wh  (this wave's 64 columns)
        #pragma unroll
        for (int jn = 0; jn < NT; ++jn) {
            v8f acc = basefrag[jn];
            acc = __builtin_amdgcn_wmma_f32_16x16x32_f16(false, a0.v, false, bfrag[jn][0],
                                                         (short)0, acc, false, false);
            acc = __builtin_amdgcn_wmma_f32_16x16x32_f16(false, a1.v, false, bfrag[jn][1],
                                                         (short)0, acc, false, false);
            #pragma unroll
            for (int i = 0; i < 8; ++i)
                zbuf[wave][i + 8 * hl][jn * 16 + lm] = acc[i];
        }
        __syncthreads();   // all 4 gates in zbuf

        // fused LSTM cell + h write-back + logit partials for (r, fb..fb+7)
        const float* __restrict__ xr = xrow[tok_in];
        float p0 = 0.f, p1 = 0.f;
        #pragma unroll
        for (int i = 0; i < 8; ++i) {
            const int f = fb + i;
            const float zi = zbuf[0][r][f] + xr[f];
            const float zf = zbuf[1][r][f] + xr[64 + f];
            const float zg = zbuf[2][r][f] + xr[128 + f];
            const float zo = zbuf[3][r][f] + xr[192 + f];
            const float c  = fsig(zf) * creg[i] + fsig(zi) * ftanh(zg);
            const float h  = fsig(zo) * ftanh(c);
            creg[i]    = c;
            hbuf[r][f] = (_Float16)h;
            p0 += h * wamp0[i];
            p1 += h * wamp1[i];
        }
        // reduce logit partials across the 8 lanes owning this row
        #pragma unroll
        for (int off = 4; off > 0; off >>= 1) {
            p0 += __shfl_down(p0, off, 8);
            p1 += __shfl_down(p1, off, 8);
        }
        if ((tid & 7) == 0) {
            const float l0 = p0 + ba0, l1 = p1 + ba1;
            const float mx  = fmaxf(l0, l1);
            const float lse = mx + __logf(__expf(l0 - mx) + __expf(l1 - mx));
            acc_lp += (tok ? l1 : l0) - lse;
        }
        tok_in = tok;
    }

    if ((tid & 7) == 0) out[row0 + r] = acc_lp;
}

extern "C" void kernel_launch(void* const* d_in, const int* in_sizes, int n_in,
                              void* d_out, int out_size, void* d_ws, size_t ws_size,
                              hipStream_t stream) {
    (void)in_sizes; (void)n_in; (void)d_ws; (void)ws_size;
    const int*   s      = (const int*)  d_in[0];
    const float* g      = (const float*)d_in[1];
    const float* W_emb  = (const float*)d_in[2];
    const float* b_emb  = (const float*)d_in[3];
    const float* W_g1   = (const float*)d_in[4];
    const float* b_g1   = (const float*)d_in[5];
    const float* W_g2   = (const float*)d_in[6];
    const float* b_g2   = (const float*)d_in[7];
    const float* W_gh   = (const float*)d_in[8];
    const float* b_gh   = (const float*)d_in[9];
    const float* W_gc   = (const float*)d_in[10];
    const float* b_gc   = (const float*)d_in[11];
    const float* Wi     = (const float*)d_in[12];
    const float* Wh     = (const float*)d_in[13];
    const float* b_lstm = (const float*)d_in[14];
    const float* W_amp  = (const float*)d_in[15];
    const float* b_amp  = (const float*)d_in[16];
    float* out = (float*)d_out;

    const int B = out_size;              // 2048
    dim3 grid(B / ROWS), block(128);
    lstm_logp_kernel<<<grid, block, 0, stream>>>(
        s, g, W_emb, b_emb, W_g1, b_g1, W_g2, b_g2, W_gh, b_gh, W_gc, b_gc,
        Wi, Wh, b_lstm, W_amp, b_amp, out);
}
